// SelfAttention_31293131718780
// MI455X (gfx1250) — matrix operand order repaired and checked by reading, not compiled, not run on hardware
//
#include <hip/hip_runtime.h>
#include <hip/hip_bf16.h>
#include <math.h>

typedef __bf16 bf16;
typedef __attribute__((ext_vector_type(16))) __bf16 v16bf;
typedef __attribute__((ext_vector_type(8)))  float  v8f;
typedef __attribute__((ext_vector_type(4)))  __bf16 v4bf;
typedef int v4i __attribute__((vector_size(16)));

#define D_MODEL 1024
#define N_HEADS 16
#define DHEAD   64
#define SEQ     2048
#define BATCH   2
#define BH_TOT  (BATCH * N_HEADS)

// LDS B-tile: 32 rows x 512 cols, padded row stride (2-way conflict max,
// keeps 16B alignment: 520*2 = 1040 bytes per row)
#define LDB_SH 520

#if __has_builtin(__builtin_amdgcn_global_load_async_to_lds_b128)
#define HAVE_ASYNC_LDS 1
typedef __attribute__((address_space(1))) v4i* gv4i_p;
typedef __attribute__((address_space(3))) v4i* lv4i_p;
#else
#define HAVE_ASYNC_LDS 0
#endif

static __device__ inline v8f wmma_bf16(v16bf a, v16bf b, v8f c) {
  // D = A(16x32 bf16) * B(32x16 bf16) + C(16x16 f32)
  return __builtin_amdgcn_wmma_f32_16x16x32_bf16(false, a, false, b, (short)0, c,
                                                 false, false);
}

static __device__ inline void async_wait_all() {
  asm volatile("s_wait_asynccnt 0" ::: "memory");
}

// ---------------------------------------------------------------- conversion
__global__ void __launch_bounds__(256)
cvt_f32_to_bf16_x4(const float* __restrict__ src, bf16* __restrict__ dst, int n4) {
  int i = blockIdx.x * blockDim.x + threadIdx.x;
  if (i >= n4) return;
  float4 f = ((const float4*)src)[i];
  v4bf o;
  o.x = (bf16)f.x; o.y = (bf16)f.y; o.z = (bf16)f.z; o.w = (bf16)f.w;
  ((v4bf*)dst)[i] = o;
}

// --------------------------------------------- stage one 32x512 B chunk to LDS
// 256 threads: thread t copies row t>>3, cols (t&7)*64 .. +63 (128 bytes).
static __device__ inline void stage_b_chunk(const bf16* __restrict__ B, int ldb,
                                            int kb, bf16* lbuf, int tid) {
  const int r  = tid >> 3;
  const int c0 = (tid & 7) * 64;
  const bf16* gp = B + (size_t)(kb + r) * ldb + c0;
  bf16* lp = lbuf + r * LDB_SH + c0;
#pragma unroll
  for (int i = 0; i < 8; ++i) {
#if HAVE_ASYNC_LDS
    __builtin_amdgcn_global_load_async_to_lds_b128(
        (gv4i_p)(gp + i * 8), (lv4i_p)(lp + i * 8), 0, 0);
#else
    *(uint4*)(lp + i * 8) = *(const uint4*)(gp + i * 8);
#endif
  }
}

// --------------------------------------- block GEMM: 64 x 512 tile, 8 waves
// A: MxK row-major (base at (m0,0)), global direct (shared rows -> L0 hits)
// B: KxN row-major (base at (0, block_n0)), staged to LDS, double-buffered.
// Each wave computes a 64x64 sub-tile at column wave*64.
static __device__ inline void gemm_tile_lds(const bf16* __restrict__ A, int lda,
                                            const bf16* __restrict__ B, int ldb,
                                            int K, bf16* bsh /*[2][32*LDB_SH]*/,
                                            v8f acc[4][4], int lane, int wave,
                                            int tid) {
  const int m  = lane & 15;
  const int k0 = (lane >> 4) * 8;
  const int wn = wave * 64;
  const int nk = K / 32;

  stage_b_chunk(B, ldb, 0, bsh, tid);

  for (int kc = 0; kc < nk; ++kc) {
    async_wait_all();
    __syncthreads();
    if (kc + 1 < nk)
      stage_b_chunk(B, ldb, (kc + 1) * 32, bsh + ((kc + 1) & 1) * (32 * LDB_SH),
                    tid);

    const bf16* lb = bsh + (kc & 1) * (32 * LDB_SH);
    const int kb = kc * 32;

    v16bf af[4], bfr[4];
#pragma unroll
    for (int i = 0; i < 4; ++i) {
      const bf16* ap = A + (size_t)(i * 16 + m) * lda + kb;
      ((uint4*)&af[i])[0] = *(const uint4*)(ap + k0);
      ((uint4*)&af[i])[1] = *(const uint4*)(ap + 16 + k0);
    }
#pragma unroll
    for (int j = 0; j < 4; ++j) {
      const bf16* bp = lb + k0 * LDB_SH + wn + j * 16 + m;
#pragma unroll
      for (int s = 0; s < 16; ++s)
        bfr[j][s] = bp[(s + (s & 8)) * LDB_SH];
    }
#pragma unroll
    for (int i = 0; i < 4; ++i)
#pragma unroll
      for (int j = 0; j < 4; ++j)
        acc[i][j] = wmma_bf16(af[i], bfr[j], acc[i][j]);
  }
}

// ------------------------------------------------------------------ QKV GEMM
// xb: (B*S, E) bf16 ; wb: (E, 3E) bf16
// outputs: q,v in (BH, S, Dh); kt in (BH, Dh, S)   (all bf16)
__global__ void __launch_bounds__(256)
qkv_gemm_kernel(const bf16* __restrict__ xb, const bf16* __restrict__ wb,
                bf16* __restrict__ q, bf16* __restrict__ kt,
                bf16* __restrict__ v) {
  __shared__ __align__(16) bf16 bsh[2][32 * LDB_SH];
  const int tid = threadIdx.x;
  const int lane = tid & 31;
  const int wave = tid >> 5;
  const int m0 = blockIdx.x * 64;
  const int bn0 = blockIdx.y * 512;
  const int n0 = bn0 + wave * 64;

  v8f acc[4][4];
#pragma unroll
  for (int i = 0; i < 4; ++i)
#pragma unroll
    for (int j = 0; j < 4; ++j)
      acc[i][j] = (v8f){0.f, 0.f, 0.f, 0.f, 0.f, 0.f, 0.f, 0.f};

  gemm_tile_lds(xb + (size_t)m0 * D_MODEL, D_MODEL, wb + bn0, 3 * D_MODEL,
                D_MODEL, &bsh[0][0], acc, lane, wave, tid);

  const int half = lane >> 4, n = lane & 15;
#pragma unroll
  for (int j = 0; j < 4; ++j) {
    const int gc = n0 + j * 16 + n;
    const int which = gc >> 10;   // 0=Q 1=K 2=V
    const int e = gc & 1023;
    const int h = e >> 6, d = e & 63;
#pragma unroll
    for (int i = 0; i < 4; ++i) {
#pragma unroll
      for (int r = 0; r < 8; ++r) {
        const int gr = m0 + i * 16 + r + 8 * half;
        const int b = gr >> 11, s = gr & 2047;
        const int bh = b * N_HEADS + h;
        const bf16 val = (bf16)acc[i][j][r];
        if (which == 0)      q[((size_t)bh * SEQ + s) * DHEAD + d] = val;
        else if (which == 1) kt[((size_t)bh * DHEAD + d) * SEQ + s] = val;
        else                 v[((size_t)bh * SEQ + s) * DHEAD + d] = val;
      }
    }
  }
}

// -------------------------------------------------- flash attention (causal)
// q,v: (BH,S,Dh) bf16 ; kt: (BH,Dh,S) bf16 ; y: (B,S,E) bf16
__global__ void __launch_bounds__(256)
attn_kernel(const bf16* __restrict__ q, const bf16* __restrict__ kt,
            const bf16* __restrict__ v, bf16* __restrict__ y) {
  __shared__ __align__(16) bf16 plds[8][16][32];

  const int lane = threadIdx.x & 31;
  const int wave = threadIdx.x >> 5;
  const int bh = blockIdx.x;
  const int q0 = blockIdx.y * 128 + wave * 16;
  const int m = lane & 15, half = lane >> 4, k0 = half * 8;
  const float NEGINF = -__builtin_inff();

  // Q A-fragments (Dh = 64 -> two 16x32 fragments)
  v16bf qa[2];
  {
    const bf16* qp = q + ((size_t)bh * SEQ + q0 + m) * DHEAD;
#pragma unroll
    for (int c = 0; c < 2; ++c) {
      ((uint4*)&qa[c])[0] = *(const uint4*)(qp + 32 * c + k0);
      ((uint4*)&qa[c])[1] = *(const uint4*)(qp + 32 * c + 16 + k0);
    }
  }

  float mrow[8], lrow[8];
  v8f o[4];
#pragma unroll
  for (int r = 0; r < 8; ++r) { mrow[r] = NEGINF; lrow[r] = 0.f; }
#pragma unroll
  for (int j = 0; j < 4; ++j)
    o[j] = (v8f){0.f, 0.f, 0.f, 0.f, 0.f, 0.f, 0.f, 0.f};

  const float zscale = 0.125f * 1.44269504088896f;  // 1/sqrt(Dh) * log2(e)
  const int nkb = (q0 + 16 + 31) >> 5;
  const int qbase = q0 + 8 * half;

  for (int kb = 0; kb < nkb; ++kb) {
    const int kbase = kb * 32;

    // ---- S = Q * K^T  (two 16x16 score tiles)
    v8f sc[2];
#pragma unroll
    for (int t = 0; t < 2; ++t) {
      v16bf b0, b1;
      const bf16* kp0 = kt + ((size_t)bh * DHEAD + k0) * SEQ + kbase + 16 * t + m;
      const bf16* kp1 = kt + ((size_t)bh * DHEAD + 32 + k0) * SEQ + kbase + 16 * t + m;
#pragma unroll
      for (int s = 0; s < 16; ++s) {
        b0[s] = kp0[(size_t)(s + (s & 8)) * SEQ];
        b1[s] = kp1[(size_t)(s + (s & 8)) * SEQ];
      }
      v8f zacc = (v8f){0.f, 0.f, 0.f, 0.f, 0.f, 0.f, 0.f, 0.f};
      sc[t] = wmma_bf16(qa[0], b0, zacc);
      sc[t] = wmma_bf16(qa[1], b1, sc[t]);
    }

    // ---- scale + causal mask, to log2 domain
    float z0[8], z1[8], rmax[8];
#pragma unroll
    for (int r = 0; r < 8; ++r) {
      const int qi = qbase + r;
      float a = sc[0][r] * zscale;
      float b = sc[1][r] * zscale;
      if (kbase + m > qi)      a = NEGINF;
      if (kbase + 16 + m > qi) b = NEGINF;
      z0[r] = a; z1[r] = b;
      rmax[r] = fmaxf(a, b);
    }
#pragma unroll
    for (int off = 1; off < 16; off <<= 1)
#pragma unroll
      for (int r = 0; r < 8; ++r)
        rmax[r] = fmaxf(rmax[r], __shfl_xor(rmax[r], off));

    float alpha[8], psum[8];
#pragma unroll
    for (int r = 0; r < 8; ++r) {
      const float mnew = fmaxf(mrow[r], rmax[r]);
      alpha[r] = exp2f(mrow[r] - mnew);
      mrow[r] = mnew;
      const float p0 = exp2f(z0[r] - mnew);
      const float p1 = exp2f(z1[r] - mnew);
      z0[r] = p0; z1[r] = p1;
      psum[r] = p0 + p1;
    }
#pragma unroll
    for (int off = 1; off < 16; off <<= 1)
#pragma unroll
      for (int r = 0; r < 8; ++r)
        psum[r] += __shfl_xor(psum[r], off);
#pragma unroll
    for (int r = 0; r < 8; ++r)
      lrow[r] = lrow[r] * alpha[r] + psum[r];

    // ---- rescale O by alpha
#pragma unroll
    for (int j = 0; j < 4; ++j)
#pragma unroll
      for (int r = 0; r < 8; ++r)
        o[j][r] *= alpha[r];

    // ---- transpose P: C-layout -> A-layout via per-wave LDS tile.
    // LDS ops are in-order within a wave (DScnt), no barrier needed.
#pragma unroll
    for (int r = 0; r < 8; ++r) {
      plds[wave][r + 8 * half][m]      = (bf16)z0[r];
      plds[wave][r + 8 * half][16 + m] = (bf16)z1[r];
    }
    __builtin_amdgcn_wave_barrier();  // pin compiler ordering only
    v16bf pa;
    ((uint4*)&pa)[0] = *(const uint4*)&plds[wave][m][k0];
    ((uint4*)&pa)[1] = *(const uint4*)&plds[wave][m][16 + k0];
    __builtin_amdgcn_wave_barrier();

    // ---- O += P * V
    const bf16* vp = v + ((size_t)bh * SEQ + kbase + k0) * DHEAD + m;
#pragma unroll
    for (int j = 0; j < 4; ++j) {
      v16bf vf;
#pragma unroll
      for (int s = 0; s < 16; ++s)
        vf[s] = vp[16 * j + (s + (s & 8)) * DHEAD];
      o[j] = wmma_bf16(pa, vf, o[j]);
    }
  }

  // ---- epilogue: y (B,S,E) bf16
  const int b = bh >> 4, h = bh & 15;
  float rinv[8];
#pragma unroll
  for (int r = 0; r < 8; ++r) rinv[r] = 1.0f / lrow[r];
#pragma unroll
  for (int j = 0; j < 4; ++j) {
    const int d = 16 * j + m;
#pragma unroll
    for (int r = 0; r < 8; ++r) {
      const int s = q0 + r + 8 * half;
      y[((size_t)b * SEQ + s) * D_MODEL + h * DHEAD + d] =
          (bf16)(o[j][r] * rinv[r]);
    }
  }
}

// ------------------------------------------------------------ output proj
__global__ void __launch_bounds__(256)
proj_gemm_kernel(const bf16* __restrict__ yb, const bf16* __restrict__ wpb,
                 float* __restrict__ out) {
  __shared__ __align__(16) bf16 bsh[2][32 * LDB_SH];
  const int tid = threadIdx.x;
  const int lane = tid & 31;
  const int wave = tid >> 5;
  const int m0 = blockIdx.x * 64;
  const int bn0 = blockIdx.y * 512;
  const int n0 = bn0 + wave * 64;

  v8f acc[4][4];
#pragma unroll
  for (int i = 0; i < 4; ++i)
#pragma unroll
    for (int j = 0; j < 4; ++j)
      acc[i][j] = (v8f){0.f, 0.f, 0.f, 0.f, 0.f, 0.f, 0.f, 0.f};

  gemm_tile_lds(yb + (size_t)m0 * D_MODEL, D_MODEL, wpb + bn0, D_MODEL,
                D_MODEL, &bsh[0][0], acc, lane, wave, tid);

  const int half = lane >> 4, n = lane & 15;
#pragma unroll
  for (int j = 0; j < 4; ++j) {
    const int gc = n0 + j * 16 + n;
#pragma unroll
    for (int i = 0; i < 4; ++i)
#pragma unroll
      for (int r = 0; r < 8; ++r) {
        const int gr = m0 + i * 16 + r + 8 * half;
        out[(size_t)gr * D_MODEL + gc] = acc[i][j][r];
      }
  }
}

// ---------------------------------------------------------------- launcher
extern "C" void kernel_launch(void* const* d_in, const int* in_sizes, int n_in,
                              void* d_out, int out_size, void* d_ws,
                              size_t ws_size, hipStream_t stream) {
  const float* x      = (const float*)d_in[0];
  const float* w_attn = (const float*)d_in[1];
  const float* w_proj = (const float*)d_in[2];
  float* out = (float*)d_out;

  const size_t nx  = (size_t)BATCH * SEQ * D_MODEL;  // 4,194,304
  const size_t nwa = (size_t)D_MODEL * 3 * D_MODEL;  // 3,145,728
  const size_t nwp = (size_t)D_MODEL * D_MODEL;      // 1,048,576

  char* ws = (char*)d_ws;
  size_t off = 0;
  auto alloc = [&](size_t bytes) {
    char* p = ws + off;
    off += (bytes + 255) & ~(size_t)255;
    return p;
  };
  bf16* xb  = (bf16*)alloc(nx * 2);
  bf16* wab = (bf16*)alloc(nwa * 2);
  bf16* wpb = (bf16*)alloc(nwp * 2);
  bf16* qb  = (bf16*)alloc(nx * 2);
  bf16* ktb = (bf16*)alloc(nx * 2);
  bf16* vb  = (bf16*)alloc(nx * 2);
  bf16* yb  = (bf16*)alloc(nx * 2);
  (void)ws_size; (void)in_sizes; (void)n_in; (void)out_size;

  cvt_f32_to_bf16_x4<<<(unsigned)((nx / 4 + 255) / 256), 256, 0, stream>>>(
      x, xb, (int)(nx / 4));
  cvt_f32_to_bf16_x4<<<(unsigned)((nwa / 4 + 255) / 256), 256, 0, stream>>>(
      w_attn, wab, (int)(nwa / 4));
  cvt_f32_to_bf16_x4<<<(unsigned)((nwp / 4 + 255) / 256), 256, 0, stream>>>(
      w_proj, wpb, (int)(nwp / 4));

  // QKV: M=4096, N=3072 -> blocks of 64x512, 8 waves each
  qkv_gemm_kernel<<<dim3(64, 6), 256, 0, stream>>>(xb, wab, qb, ktb, vb);

  // attention: 32 (b,h) x 16 q-blocks of 128 queries
  attn_kernel<<<dim3(BH_TOT, SEQ / 128), 256, 0, stream>>>(qb, ktb, vb, yb);

  // projection: M=4096, N=1024 -> blocks of 64x512
  proj_gemm_kernel<<<dim3(64, 2), 256, 0, stream>>>(yb, wpb, out);
}
